// KGANEncoder_9277129359390
// MI455X (gfx1250) — compile-verified, matrix operand order checked by reading, not run on hardware
//
#include <hip/hip_runtime.h>

typedef __attribute__((ext_vector_type(16))) _Float16     v16h;
typedef __attribute__((ext_vector_type(8)))  float        v8f;
typedef __attribute__((ext_vector_type(4)))  unsigned int v4u;
typedef __attribute__((ext_vector_type(8)))  int          v8i;
typedef __attribute__((ext_vector_type(4)))  int          v4i;

// Pack 16 halves for a WMMA 16x16x32 f16 operand from two 8-float contiguous
// chunks (per-lane layout: halves 0..7 -> K=k0+8g.. , halves 8..15 -> K=k0+16+8g..)
__device__ inline v16h pack16_from_f32(const float* __restrict__ p0,
                                       const float* __restrict__ p1) {
  const float4* a = (const float4*)p0;
  const float4* b = (const float4*)p1;
  float4 x0 = a[0], x1 = a[1], y0 = b[0], y1 = b[1];
  v16h r;
  r[0]=(_Float16)x0.x;  r[1]=(_Float16)x0.y;  r[2]=(_Float16)x0.z;  r[3]=(_Float16)x0.w;
  r[4]=(_Float16)x1.x;  r[5]=(_Float16)x1.y;  r[6]=(_Float16)x1.z;  r[7]=(_Float16)x1.w;
  r[8]=(_Float16)y0.x;  r[9]=(_Float16)y0.y;  r[10]=(_Float16)y0.z; r[11]=(_Float16)y0.w;
  r[12]=(_Float16)y1.x; r[13]=(_Float16)y1.y; r[14]=(_Float16)y1.z; r[15]=(_Float16)y1.w;
  return r;
}

// Norm-clamp rows of a (nrows x 64) table: row *= (|row|>1 ? 1/(|row|+1e-7) : 1)
__global__ __launch_bounds__(256) void norm_kernel(const float* __restrict__ src,
                                                   float* __restrict__ dst, int nrows) {
  int wave = threadIdx.x >> 5, lane = threadIdx.x & 31;
  int row = blockIdx.x * 8 + wave;
  if (row >= nrows) return;
  long base = (long)row * 64;
  float f0 = src[base + lane];
  float f1 = src[base + 32 + lane];
  float ss = f0 * f0 + f1 * f1;
#pragma unroll
  for (int m = 16; m >= 1; m >>= 1) ss += __shfl_xor(ss, m, 32);
  float nrm = sqrtf(ss);
  float sc = nrm > 1.f ? 1.f / (nrm + 1e-7f) : 1.f;
  dst[base + lane] = f0 * sc;
  dst[base + 32 + lane] = f1 * sc;
}

// Per seed b: Xc[b][0:64] = h = En[entity_idx[b]];  Xc[1024+b][0:64] = hsum = sum_k En[ent1[b,k]]
__global__ __launch_bounds__(256) void gather_kernel(const int* __restrict__ entity_idx,
                                                     const int* __restrict__ adj_e,
                                                     const float* __restrict__ En,
                                                     float* __restrict__ Xc) {
  __shared__ float red[256];
  int b = blockIdx.x, tid = threadIdx.x;
  int e0 = entity_idx[b];
  if (tid < 64) Xc[b * 128 + tid] = En[(long)e0 * 64 + tid];
  int e = tid & 63, q = tid >> 6;
  float acc = 0.f;
  for (int k = q; k < 32; k += 4) acc += En[(long)adj_e[e0 * 32 + k] * 64 + e];
  red[tid] = acc;
  __syncthreads();
  if (q == 0)
    Xc[(1024 + b) * 128 + e] = red[e] + red[64 + e] + red[128 + e] + red[192 + e];
}

// C[M x 64] = act( A[M x KDIM] @ W[64 x KDIM]^T + bias ), f16 WMMA, one 16x16 tile per wave.
// ACT: 0 = none, 1 = leaky(0.2).  KDIM templated so the K loop fully unrolls into
// back-to-back v_wmma with cvt VALU work overlapped.
template <int KDIM, int ACT>
__global__ __launch_bounds__(32) void gemm_n64_wmma(const float* __restrict__ A, int lda,
                                                    const float* __restrict__ W, int ldw,
                                                    const float* __restrict__ bias,
                                                    float* __restrict__ C, int ldc) {
  int tile = blockIdx.x;
  int mt = tile >> 2, nt = tile & 3;
  int lane = threadIdx.x;
  int g = lane >> 4, l16 = lane & 15;
  int arow = mt * 16 + l16;
  int wrow = nt * 16 + l16;
  v8f acc = {};
#pragma unroll
  for (int k0 = 0; k0 < KDIM; k0 += 32) {
    const float* ap = A + (long)arow * lda + k0 + 8 * g;
    const float* wp = W + (long)wrow * ldw + k0 + 8 * g;
    v16h av = pack16_from_f32(ap, ap + 16);
    v16h bv = pack16_from_f32(wp, wp + 16);
    acc = __builtin_amdgcn_wmma_f32_16x16x32_f16(false, av, false, bv, (short)0, acc,
                                                 false, false);
  }
  int nn = nt * 16 + l16;
  float bb = bias ? bias[nn] : 0.f;
#pragma unroll
  for (int v = 0; v < 8; ++v) {
    int m = mt * 16 + v + 8 * g;
    float val = acc[v] + bb;
    if (ACT == 1) val = val >= 0.f ? val : 0.2f * val;
    C[(long)m * ldc + nn] = val;
  }
}

// Per (hop,b): X = relu(Q[sb] + P) (64 rels x 64), Y = relu(X @ w2^T), s = sigmoid(Y . w3)
// w2 (64x64 f32, 16KB) is staged into LDS once per workgroup via the Tensor Data Mover.
__global__ __launch_bounds__(256) void score_kernel(const float* __restrict__ Q,
                                                    const float* __restrict__ P,
                                                    const float* __restrict__ w2,
                                                    const float* __restrict__ w3,
                                                    float* __restrict__ S) {
  __shared__ float    Ws[64 * 64];   // staged att_w2 (TDM)
  __shared__ _Float16 Xs[64 * 64];
  __shared__ float    Ys[64 * 64];
  int sb = blockIdx.x;
  int tid = threadIdx.x;
  int wave = tid >> 5, lane = tid & 31;

  if (wave == 0) {
    // Build Tensor DMA Descriptor (D#) for a 64x64 f32 tile, contiguous stride 64.
    unsigned lds_off = (unsigned)(unsigned long long)(const void*)&Ws[0];
    unsigned long long ga = (unsigned long long)(const void*)w2;
    v4u g0;
    g0[0] = 0x1u;                                   // count=1, user descriptor
    g0[1] = lds_off;                                // lds_addr (bytes)
    g0[2] = (unsigned)(ga & 0xFFFFFFFFu);           // global_addr[31:0]
    g0[3] = (unsigned)((ga >> 32) & 0x01FFFFFFu)    // global_addr[56:32]
            | (2u << 30);                           // type = 2 (image)
    v8i g1;
    g1[0] = (int)(2u << 16);        // data_size = 2 (4 bytes); no mask/pad/iterate
    g1[1] = (int)(64u << 16);       // tensor_dim0[15:0] = 64 (bits 63:48 of group)
    g1[2] = (int)(64u << 16);       // tensor_dim0 hi=0; tensor_dim1[15:0] = 64
    g1[3] = (int)(64u << 16);       // tensor_dim1 hi=0; tile_dim0 = 64
    g1[4] = 64;                     // tile_dim1 = 64; tile_dim2 = 0 (2D)
    g1[5] = 64;                     // tensor_dim0_stride[31:0] = 64
    g1[6] = 0;                      // stride hi / dim1_stride (unused for 2D tile)
    g1[7] = 0;
    v4i gz4 = {0, 0, 0, 0};         // groups 2/3 unused for 2D tensors
    v8i gz8 = {0, 0, 0, 0, 0, 0, 0, 0};
    __builtin_amdgcn_tensor_load_to_lds(g0, g1, gz4, gz4, gz8, 0);
    __builtin_amdgcn_s_wait_tensorcnt(0);
  }

  const float* q = Q + sb * 64;
  for (int idx = tid; idx < 4096; idx += 256) {
    int rel = idx >> 6, e = idx & 63;
    float x = q[e] + P[rel * 64 + e];
    Xs[idx] = (_Float16)fmaxf(x, 0.f);
  }
  __syncthreads();

  int g = lane >> 4, l16 = lane & 15;
#pragma unroll
  for (int ti = 0; ti < 2; ++ti) {
    int t = wave + ti * 8;
    int mt = t >> 2, nt = t & 3;
    int arow = mt * 16 + l16;
    int wrow = nt * 16 + l16;
    v8f acc = {};
#pragma unroll
    for (int ks = 0; ks < 2; ++ks) {
      int k0 = ks * 32;
      const _Float16* xp = &Xs[arow * 64 + k0 + 8 * g];
      v16h av;
#pragma unroll
      for (int j = 0; j < 8; ++j) { av[j] = xp[j]; av[8 + j] = xp[16 + j]; }
      const float* wp = &Ws[wrow * 64 + k0 + 8 * g];   // LDS-staged weights
      v16h bv = pack16_from_f32(wp, wp + 16);
      acc = __builtin_amdgcn_wmma_f32_16x16x32_f16(false, av, false, bv, (short)0, acc,
                                                   false, false);
    }
#pragma unroll
    for (int v = 0; v < 8; ++v) {
      int m = mt * 16 + v + 8 * g;
      int nn = nt * 16 + l16;
      Ys[m * 64 + nn] = fmaxf(acc[v], 0.f);  // relu fused
    }
  }
  __syncthreads();
  int r = tid >> 2, p = tid & 3;
  float sum = 0.f;
  const float* yr = &Ys[r * 64 + p * 16];
#pragma unroll
  for (int j = 0; j < 16; ++j) sum += yr[j] * w3[p * 16 + j];
  sum += __shfl_xor(sum, 1, 32);
  sum += __shfl_xor(sum, 2, 32);
  if (p == 0) S[sb * 64 + r] = 1.f / (1.f + expf(-sum));
}

// Per (hop,b): softmax over neighbor scores (via relation lookup) + weighted gather of En rows
__global__ __launch_bounds__(256) void agg_kernel(const int* __restrict__ entity_idx,
                                                  const int* __restrict__ adj_e,
                                                  const int* __restrict__ adj_r,
                                                  const float* __restrict__ En,
                                                  const float* __restrict__ S,
                                                  float* __restrict__ Agg) {
  __shared__ float srow[64];
  __shared__ int ent[1024];
  __shared__ float wts[1024];
  __shared__ float red[256];
  int blk = blockIdx.x;
  int sIdx = blk >> 10;  // 0 = hop1 (32 nbrs), 1 = hop2 (1024 nbrs)
  int b = blk & 1023;
  int tid = threadIdx.x;
  if (tid < 64) srow[tid] = S[blk * 64 + tid];
  __syncthreads();
  int e0 = entity_idx[b];
  int N = sIdx ? 1024 : 32;
  for (int n = tid; n < N; n += 256) {
    int src = sIdx ? adj_e[e0 * 32 + (n >> 5)] : e0;
    int i = sIdx ? (n & 31) : n;
    ent[n] = adj_e[src * 32 + i];
    wts[n] = srow[adj_r[src * 32 + i]];
  }
  __syncthreads();
  float lmax = -1e30f;
  for (int n = tid; n < N; n += 256) lmax = fmaxf(lmax, wts[n]);
  red[tid] = lmax;
  __syncthreads();
  for (int s = 128; s > 0; s >>= 1) {
    if (tid < s) red[tid] = fmaxf(red[tid], red[tid + s]);
    __syncthreads();
  }
  float amax = red[0];
  __syncthreads();
  float lsum = 0.f;
  for (int n = tid; n < N; n += 256) {
    float ev = expf(wts[n] - amax);
    wts[n] = ev;
    lsum += ev;
  }
  red[tid] = lsum;
  __syncthreads();
  for (int s = 128; s > 0; s >>= 1) {
    if (tid < s) red[tid] += red[tid + s];
    __syncthreads();
  }
  float inv = 1.f / red[0];
  __syncthreads();
  int e = tid & 63, q = tid >> 6;
  float acc = 0.f;
  for (int n = q; n < N; n += 4) {
    if (n + 16 < N)  // hide L2 hit latency on the random row gather
      __builtin_prefetch(&En[(long)ent[n + 16] * 64 + e], 0, 1);
    acc += wts[n] * En[(long)ent[n] * 64 + e];
  }
  red[tid] = acc;
  __syncthreads();
  if (q == 0)
    Agg[blk * 64 + e] = (red[e] + red[64 + e] + red[128 + e] + red[192 + e]) * inv;
}

// out[b] = [emb2 | emb1 | h]
__global__ void out_kernel(const float* __restrict__ Emb, const float* __restrict__ Xc,
                           float* __restrict__ out) {
  int i = blockIdx.x * 256 + threadIdx.x;
  if (i >= 1024 * 192) return;
  int b = i / 192, j = i % 192;
  float v;
  if (j < 64)        v = Emb[(1024 + b) * 64 + j];        // emb2
  else if (j < 128)  v = Emb[b * 64 + (j - 64)];          // emb1
  else               v = Xc[b * 128 + (j - 128)];         // h
  out[i] = v;
}

extern "C" void kernel_launch(void* const* d_in, const int* in_sizes, int n_in,
                              void* d_out, int out_size, void* d_ws, size_t ws_size,
                              hipStream_t stream) {
  const int*   entity_idx = (const int*)d_in[0];
  const int*   adj_e      = (const int*)d_in[1];
  const int*   adj_r      = (const int*)d_in[2];
  const float* E          = (const float*)d_in[3];
  const float* R          = (const float*)d_in[4];
  const float* w1         = (const float*)d_in[5];   // (64,128): w1h = cols 0..63, w1r = cols 64..127
  const float* w2         = (const float*)d_in[6];   // (64,64)
  const float* w3         = (const float*)d_in[7];   // (64,)
  const float* wx         = (const float*)d_in[8];   // (64,64)
  const float* wxb        = (const float*)d_in[9];   // (64,)
  const float* wc         = (const float*)d_in[10];  // (64,128)
  const float* wcb        = (const float*)d_in[11];  // (64,)

  char* ws = (char*)d_ws;
  float* En  = (float*)(ws);                 // 100000*64*4 = 25,600,000 B
  float* Rn  = (float*)(ws + 25600000);      // 16 KB
  float* P   = (float*)(ws + 25616384);      // 16 KB  (Rn @ w1r^T)
  float* Xc  = (float*)(ws + 25632768);      // 2048 x 128 f32 = 1 MB  ([h;hsum | v1;v2])
  float* Q   = (float*)(ws + 26681344);      // 2048 x 64 ([h;hsum] @ w1h^T)
  float* S   = (float*)(ws + 27205632);      // 2048 x 64 per-relation attention scores
  float* Agg = (float*)(ws + 27729920);      // 2048 x 64 softmax-weighted neighbor agg
  float* Emb = (float*)(ws + 28254208);      // 2048 x 64 [emb1;emb2]

  norm_kernel<<<12500, 256, 0, stream>>>(E, En, 100000);
  norm_kernel<<<8, 256, 0, stream>>>(R, Rn, 64);
  gather_kernel<<<1024, 256, 0, stream>>>(entity_idx, adj_e, En, Xc);
  // P = Rn @ w1r^T   (64x64, K=64)
  gemm_n64_wmma<64, 0><<<16, 32, 0, stream>>>(Rn, 64, w1 + 64, 128, nullptr, P, 64);
  // Q = [h; hsum] @ w1h^T   (2048x64, K=64)
  gemm_n64_wmma<64, 0><<<512, 32, 0, stream>>>(Xc, 128, w1, 128, nullptr, Q, 64);
  score_kernel<<<2048, 256, 0, stream>>>(Q, P, w2, w3, S);
  agg_kernel<<<2048, 256, 0, stream>>>(entity_idx, adj_e, adj_r, En, S, Agg);
  // V = leaky(Agg @ wx^T + wxb) -> Xc cols 64..127
  gemm_n64_wmma<64, 1><<<512, 32, 0, stream>>>(Agg, 64, wx, 64, wxb, Xc + 64, 128);
  // Emb = leaky([x | V] @ wc^T + wcb)  (K=128)
  gemm_n64_wmma<128, 1><<<512, 32, 0, stream>>>(Xc, 128, wc, 128, wcb, Emb, 64);
  out_kernel<<<768, 256, 0, stream>>>(Emb, Xc, (float*)d_out);
}
